// BoltzmannMachine_76536317215384
// MI455X (gfx1250) — matrix operand order repaired and checked by reading, not compiled, not run on hardware
//
#include <hip/hip_runtime.h>
#include <hip/hip_bf16.h>
#include <math.h>

// ---------------------------------------------------------------- constants
#define BATCH 64
#define VS    4096
#define HS    2048
#define KTOT  (HS + VS)              // 6144 = unified K (hid cols 0..2047, vis cols 2048..6143)
#define KT_TILES (KTOT / 32)         // 192 k-tiles of 32
#define SPLITK 8
#define KT_PER_SPLIT (KT_TILES / SPLITK) // 24 k-tiles per split
#define STEPS 100

// A k-slice staged in LDS: 64 rows x (24*32) cols bf16 = 98304 bytes
#define A_SLICE_BYTES (BATCH * KT_PER_SPLIT * 32 * 2)
#define A_CHUNKS      (A_SLICE_BYTES / 16)            // 6144 16B chunks
#define CHUNKS_PER_THR (A_CHUNKS / 256)               // 24

typedef __attribute__((ext_vector_type(16))) __bf16 bf16x16;
typedef __attribute__((ext_vector_type(8)))  float  f32x8;
typedef unsigned short u16;
typedef unsigned int   u32;

// workspace layout (bytes)
#define WH_OFF 0ull                                   // 6144*2048 bf16 = 24 MB
#define WV_OFF (WH_OFF + (size_t)KTOT * HS * 2)       // 6144*4096 bf16 = 48 MB
#define SF_OFF (WV_OFF + (size_t)KTOT * VS * 2)       // 64*6144 f32
#define SB_OFF (SF_OFF + (size_t)BATCH * KTOT * 4)    // 64*6144 bf16
#define P_OFF  (SB_OFF + (size_t)BATCH * KTOT * 2)    // SPLITK*64*4096 f32 = 8 MB (max N)

// ---------------------------------------------------------------- helpers
__device__ __forceinline__ u16 f2bf(float f) {
    u32 u = __float_as_uint(f);
    u32 r = u + 0x7FFFu + ((u >> 16) & 1u);   // round-to-nearest-even
    return (u16)(r >> 16);
}

struct Frag32B { float4 lo, hi; };

__device__ __forceinline__ bf16x16 load_frag(const u16* p) {
    Frag32B t;
    t.lo = *(const float4*)(p);        // 16B aligned
    t.hi = *(const float4*)(p + 8);    // 16B aligned
    return __builtin_bit_cast(bf16x16, t);
}

#define WMMA_BF16(A, B, C) \
    __builtin_amdgcn_wmma_f32_16x16x32_bf16(false, (A), false, (B), (short)0, (C), false, false)

// B-fragment element offset (u16 units) for W element (k, n):
// tile (kt=k/32, nt=n/16), lane = (n&15) | ((k>>4)&1)<<4, elem = k&15
__device__ __forceinline__ size_t wfrag_idx(int k, int n) {
    int lane = (n & 15) | (((k >> 4) & 1) << 4);
    return ((size_t)(n >> 4) * KT_TILES + (size_t)(k >> 5)) * 512
         + (size_t)lane * 16 + (size_t)(k & 15);
}

// ---------------------------------------------------------------- weight builds
// W_h (6144 x 2048): rows 0..2047 = symmetrized hh (triu+triu^T, zero diag),
//                    rows 2048..  = vis_hid
__global__ void build_wh(const float* __restrict__ hh_raw,
                         const float* __restrict__ vis_hid,
                         u16* __restrict__ W) {
    long long idx = (long long)blockIdx.x * blockDim.x + threadIdx.x;
    if (idx >= (long long)KTOT * HS) return;
    int k = (int)(idx / HS), n = (int)(idx % HS);
    float v;
    if (k < HS) {
        v = (k < n) ? hh_raw[(size_t)k * HS + n]
          : (k > n) ? hh_raw[(size_t)n * HS + k] : 0.0f;
    } else {
        v = vis_hid[(size_t)(k - HS) * HS + n];
    }
    W[wfrag_idx(k, n)] = f2bf(v);
}

// W_v (6144 x 4096): rows 0..2047 = vis_hid^T, rows 2048.. = symmetrized vv
__global__ void build_wv(const float* __restrict__ vv_raw,
                         const float* __restrict__ vis_hid,
                         u16* __restrict__ W) {
    long long idx = (long long)blockIdx.x * blockDim.x + threadIdx.x;
    if (idx >= (long long)KTOT * VS) return;
    int k = (int)(idx / VS), n = (int)(idx % VS);
    float v;
    if (k < HS) {
        v = vis_hid[(size_t)n * HS + k];            // vis_hid^T
    } else {
        int i = k - HS, j = n;
        v = (i < j) ? vv_raw[(size_t)i * VS + j]
          : (i > j) ? vv_raw[(size_t)j * VS + i] : 0.0f;
    }
    W[wfrag_idx(k, n)] = f2bf(v);
}

// ---------------------------------------------------------------- state init
__global__ void init_state(const float* __restrict__ x,
                           float* __restrict__ Sf, u16* __restrict__ Sb) {
    int idx = blockIdx.x * blockDim.x + threadIdx.x;   // BATCH*KTOT = 393216
    int m = idx / KTOT, c = idx % KTOT;
    float v = (c < HS) ? 0.5f : x[(size_t)m * VS + (c - HS)];
    Sf[idx] = v;
    Sb[idx] = f2bf(v);
}

// ---------------------------------------------------------------- split-K GEMM
// grid: (N/128, SPLITK), block 256 (8 waves). Wave computes a 64x16 strip.
// A k-slice (64 x 768 bf16) is async-staged into LDS in fragment-major tile
// layout: tile (kt, mt) at byte (kt*4+mt)*1024, within-tile lane*32 bytes.
// Both A (LDS) and B (global/L2) fragments are software-pipelined one
// iteration ahead so WMMAs never wait on a full counter drain.
__global__ __launch_bounds__(256) void gemm_splitk(
        const u16* __restrict__ Sb,   // A: 64 x 6144 bf16, row-major
        const u16* __restrict__ W,    // B: fragment-major tiles
        float* __restrict__ P,        // partials [SPLITK][64][N]
        int N) {
    extern __shared__ u16 ldsA[];     // A_SLICE_BYTES dynamic LDS

    const int tid   = threadIdx.x;
    const int lane  = tid & 31;
    const int wave  = tid >> 5;
    const int nt    = blockIdx.x * 8 + wave;
    const int ks    = blockIdx.y;
    const int kt0   = ks * KT_PER_SPLIT;

    // ---- async-stage A slice into LDS (fragment-major) -------------------
    // chunk = ((kt*4 + mt)*32 + frag_lane)*2 + half  (16 bytes each)
    #pragma unroll
    for (int c = 0; c < CHUNKS_PER_THR; ++c) {
        int chunk  = tid + c * 256;
        int half   = chunk & 1;
        int flane  = (chunk >> 1) & 31;
        int tileid = chunk >> 6;          // kt*4 + mt
        int mt     = tileid & 3;
        int kt     = tileid >> 2;
        int row    = mt * 16 + (flane & 15);
        int kcol   = (kt0 + kt) * 32 + ((flane >> 4) << 4) + half * 8;
        const u16* g = Sb + (size_t)row * KTOT + kcol;
        u32 ldsoff = (u32)chunk * 16;
        asm volatile("global_load_async_to_lds_b128 %0, %1, off"
                     :: "v"(ldsoff), "v"(g) : "memory");
    }
    asm volatile("s_wait_asynccnt 0x0" ::: "memory");
    __syncthreads();

    // ---- main loop: everything prefetched one iteration ahead ------------
    const u16* wp = W + ((size_t)nt * KT_TILES + kt0) * 512 + (size_t)lane * 16;
    const u16* la = ldsA + (size_t)lane * 16;   // + (kt*4+mt)*512 per tile

    f32x8 acc0 = {0,0,0,0,0,0,0,0};
    f32x8 acc1 = {0,0,0,0,0,0,0,0};
    f32x8 acc2 = {0,0,0,0,0,0,0,0};
    f32x8 acc3 = {0,0,0,0,0,0,0,0};

    // preload iteration 0
    bf16x16 bc  = load_frag(wp);
    bf16x16 a0c = load_frag(la + 0 * 512);
    bf16x16 a1c = load_frag(la + 1 * 512);
    bf16x16 a2c = load_frag(la + 2 * 512);
    bf16x16 a3c = load_frag(la + 3 * 512);

    #pragma unroll 2
    for (int kt = 0; kt < KT_PER_SPLIT; ++kt) {
        // prefetch next iteration's fragments while WMMAs consume current.
        // One-past-end B read stays inside workspace; one-past-end LDS read
        // returns zero per ISA OOB rule. Neither value is ever consumed.
        bf16x16 bn = load_frag(wp + 512);
        const u16* lan = la + (size_t)((kt + 1) * 4) * 512;
        bf16x16 a0n = load_frag(lan + 0 * 512);
        bf16x16 a1n = load_frag(lan + 1 * 512);
        bf16x16 a2n = load_frag(lan + 2 * 512);
        bf16x16 a3n = load_frag(lan + 3 * 512);
        __builtin_prefetch(wp + 4 * 512, 0, 1);

        acc0 = WMMA_BF16(a0c, bc, acc0);
        acc1 = WMMA_BF16(a1c, bc, acc1);
        acc2 = WMMA_BF16(a2c, bc, acc2);
        acc3 = WMMA_BF16(a3c, bc, acc3);

        bc = bn; a0c = a0n; a1c = a1n; a2c = a2n; a3c = a3n;
        wp += 512;
    }

    // C/D layout: lane 0-15 -> n=lane, m=r; lane 16-31 -> n=lane-16, m=r+8
    const int mhi = (lane >> 4) << 3;
    float* pb = P + (size_t)ks * BATCH * N + (size_t)(nt * 16 + (lane & 15));
    #pragma unroll
    for (int r = 0; r < 8; ++r) {
        pb[(size_t)( 0 + mhi + r) * N] = acc0[r];
        pb[(size_t)(16 + mhi + r) * N] = acc1[r];
        pb[(size_t)(32 + mhi + r) * N] = acc2[r];
        pb[(size_t)(48 + mhi + r) * N] = acc3[r];
    }
}

// ---------------------------------------------------------------- reduce + sigmoid + blend
__global__ void act_update(const float* __restrict__ P,
                           const float* __restrict__ bias,
                           float* __restrict__ Sf, u16* __restrict__ Sb,
                           int N, int colOff, float invTemp) {
    int idx = blockIdx.x * blockDim.x + threadIdx.x;   // BATCH*N
    int m = idx / N, n = idx % N;
    size_t span = (size_t)BATCH * N;
    float s = 0.0f;
    #pragma unroll
    for (int ksp = 0; ksp < SPLITK; ++ksp) s += P[(size_t)ksp * span + idx];
    s = (s + bias[n]) * invTemp;
    float prob = 1.0f / (1.0f + __expf(-s));
    size_t si = (size_t)m * KTOT + colOff + n;
    float nv = 0.9f * Sf[si] + 0.1f * prob;
    Sf[si] = nv;
    Sb[si] = f2bf(nv);
}

// ---------------------------------------------------------------- final copy
__global__ void copy_out(const float* __restrict__ Sf, float* __restrict__ out) {
    int idx = blockIdx.x * blockDim.x + threadIdx.x;   // BATCH*VS
    int m = idx / VS, n = idx % VS;
    out[idx] = Sf[(size_t)m * KTOT + HS + n];
}

// ---------------------------------------------------------------- launcher
extern "C" void kernel_launch(void* const* d_in, const int* in_sizes, int n_in,
                              void* d_out, int out_size, void* d_ws, size_t ws_size,
                              hipStream_t stream) {
    (void)in_sizes; (void)n_in; (void)out_size; (void)ws_size;

    const float* x        = (const float*)d_in[0];   // 64 x 4096
    const float* vis_bias = (const float*)d_in[1];   // 4096
    const float* hid_bias = (const float*)d_in[2];   // 2048
    const float* vis_hid  = (const float*)d_in[3];   // 4096 x 2048
    const float* vv_raw   = (const float*)d_in[4];   // 4096 x 4096
    const float* hh_raw   = (const float*)d_in[5];   // 2048 x 2048
    // d_in[6] = max_steps (always 100, compile-time STEPS)

    char*  ws = (char*)d_ws;
    u16*   Wh = (u16*)(ws + WH_OFF);
    u16*   Wv = (u16*)(ws + WV_OFF);
    float* Sf = (float*)(ws + SF_OFF);
    u16*   Sb = (u16*)(ws + SB_OFF);
    float* P  = (float*)(ws + P_OFF);

    // one-time (per launch) weight symmetrize + bf16 + WMMA-B-fragment swizzle
    build_wh<<<((size_t)KTOT * HS + 255) / 256, 256, 0, stream>>>(hh_raw, vis_hid, Wh);
    build_wv<<<((size_t)KTOT * VS + 255) / 256, 256, 0, stream>>>(vv_raw, vis_hid, Wv);
    init_state<<<(BATCH * KTOT) / 256, 256, 0, stream>>>(x, Sf, Sb);

    for (int i = 0; i < STEPS; ++i) {
        float temp = 0.01f * (1.0f + 4.0f * expf(-5.0f * (float)i / (float)STEPS));
        float invT = 1.0f / temp;

        // hid update: P = S(64x6144) . W_h(6144x2048)
        gemm_splitk<<<dim3(HS / 128, SPLITK), 256, A_SLICE_BYTES, stream>>>(Sb, Wh, P, HS);
        act_update<<<(BATCH * HS) / 256, 256, 0, stream>>>(P, hid_bias, Sf, Sb, HS, 0, invT);

        // vis update: P = S(64x6144) . W_v(6144x4096)
        gemm_splitk<<<dim3(VS / 128, SPLITK), 256, A_SLICE_BYTES, stream>>>(Sb, Wv, P, VS);
        act_update<<<(BATCH * VS) / 256, 256, 0, stream>>>(P, vis_bias, Sf, Sb, VS, HS, invT);
    }

    copy_out<<<(BATCH * VS) / 256, 256, 0, stream>>>(Sf, (float*)d_out);
}
